// RCNModel_52922587021478
// MI455X (gfx1250) — compile-verified, matrix operand order checked by reading, not run on hardware
//
#include <hip/hip_runtime.h>
#include <hip/hip_bf16.h>

#define NUM_GRAPHS 256
#define NPG 64
#define EPG 1280
#define NN (NUM_GRAPHS * NPG)   // 16384
#define HH 4
#define CC 128
#define HC 512
#define ND 64
#define ED 32
#define POLICY 4672
#define SROW 513                // LDS row stride (bank spread)

typedef __attribute__((ext_vector_type(16))) __bf16 v16bf;
typedef __attribute__((ext_vector_type(8)))  __bf16 v8bf;
typedef __attribute__((ext_vector_type(8)))  float  v8f;
typedef __attribute__((ext_vector_type(4)))  float  v4f;

// ---- monotonic float<->uint encoding for atomicMax on LDS ----
__device__ __forceinline__ unsigned fenc(float f) {
    unsigned b = __float_as_uint(f);
    return (b & 0x80000000u) ? ~b : (b | 0x80000000u);
}
__device__ __forceinline__ float fdec(unsigned u) {
    unsigned b = (u & 0x80000000u) ? (u & 0x7FFFFFFFu) : ~u;
    return __uint_as_float(b);
}

// load one 16-lane x 32-K bf16 WMMA fragment (row-major source, stride K)
__device__ __forceinline__ v16bf ldfrag(const __bf16* p) {
    v8bf lo = *(const v8bf*)p;
    v8bf hi = *(const v8bf*)(p + 16);
    v16bf r;
#pragma unroll
    for (int i = 0; i < 8; ++i) { r[i] = lo[i]; r[i + 8] = hi[i]; }
    return r;
}

// =====================================================================
// WMMA GEMM: C[M,Nc] = A[M,K](bf16) * B (given as BT[Nc,K] bf16) + bias
// One wave computes a 16x64 strip (4 accumulators): the A fragment is
// loaded once per K-step and reused across 4 B fragments (2.5 loads per
// WMMA instead of 4; 4x less A-side refetch on the big GEMMs).
// =====================================================================
__global__ __launch_bounds__(256) void gemm_bf16_wmma(
    const __bf16* __restrict__ A,    // [M,K] row-major
    const __bf16* __restrict__ BT,   // [Nc,K] row-major (i.e. B transposed)
    const float*  __restrict__ bias, // [Nc]
    float*        __restrict__ Cf,   // [M,Nc] or null
    __bf16*       __restrict__ Cbf,  // [M,Nc] or null
    int M, int Nc, int K, int act)
{
    int wave    = blockIdx.x * (blockDim.x >> 5) + (threadIdx.x >> 5);
    int stripsN = Nc >> 6;                 // 64-wide output strips
    int tm = wave / stripsN;
    int tn = wave - tm * stripsN;
    if (tm >= (M >> 4)) return;            // wave-uniform: EXEC stays all-ones

    int lane = threadIdx.x & 31;
    int half = lane >> 4;                  // 0: K 0-7/16-23, 1: K 8-15/24-31
    int idx  = lane & 15;

    const __bf16* arow  = A  + (size_t)(tm * 16 + idx) * K;
    const __bf16* brow0 = BT + (size_t)(tn * 64 +  0 + idx) * K;
    const __bf16* brow1 = BT + (size_t)(tn * 64 + 16 + idx) * K;
    const __bf16* brow2 = BT + (size_t)(tn * 64 + 32 + idx) * K;
    const __bf16* brow3 = BT + (size_t)(tn * 64 + 48 + idx) * K;

    v8f acc0 = {}, acc1 = {}, acc2 = {}, acc3 = {};
    for (int k0 = 0; k0 < K; k0 += 32) {
        int kb = k0 + half * 8;
        v16bf av = ldfrag(arow  + kb);
        v16bf b0 = ldfrag(brow0 + kb);
        v16bf b1 = ldfrag(brow1 + kb);
        v16bf b2 = ldfrag(brow2 + kb);
        v16bf b3 = ldfrag(brow3 + kb);
        acc0 = __builtin_amdgcn_wmma_f32_16x16x32_bf16(false, av, false, b0, (short)0, acc0, false, false);
        acc1 = __builtin_amdgcn_wmma_f32_16x16x32_bf16(false, av, false, b1, (short)0, acc1, false, false);
        acc2 = __builtin_amdgcn_wmma_f32_16x16x32_bf16(false, av, false, b2, (short)0, acc2, false, false);
        acc3 = __builtin_amdgcn_wmma_f32_16x16x32_bf16(false, av, false, b3, (short)0, acc3, false, false);
    }

    int row0 = tm * 16 + half * 8;
    v8f accs[4] = {acc0, acc1, acc2, acc3};
#pragma unroll
    for (int s = 0; s < 4; ++s) {
        int col  = tn * 64 + s * 16 + idx;
        float bb = bias ? bias[col] : 0.f;
#pragma unroll
        for (int v = 0; v < 8; ++v) {
            float val = accs[s][v] + bb;
            if (act) val = val > 0.f ? val : 0.f;
            size_t o = (size_t)(row0 + v) * Nc + col;
            if (Cf)  Cf[o]  = val;
            if (Cbf) Cbf[o] = (__bf16)val;
        }
    }
}

// =====================================================================
// GATv2 edge phase + bias + LayerNorm + ReLU, one workgroup per graph.
// LDS: xl/xr tiles (64x513 f32 each), per-edge alpha, per-(node,head)
// max/denom, att vector, eW table.  ~303 KB total (fits 320 KB WGP LDS).
// =====================================================================
__global__ __launch_bounds__(256) void gatv2_kernel(
    const float* __restrict__ xl, const float* __restrict__ xr,   // [N,HC]
    const float* __restrict__ eW,                                  // [8,HC]
    const int* __restrict__ esrc, const int* __restrict__ edst,
    const int* __restrict__ eattr,
    const float* __restrict__ att,                                 // [HC]
    const float* __restrict__ bias,                                // concat? HC : CC
    const float* __restrict__ lng, const float* __restrict__ lnb,
    int concat,
    float*  __restrict__ outf,   // [N,CC]  (layer 2) or null
    __bf16* __restrict__ outbf)  // [N,HC]  (layer 1) or null
{
    extern __shared__ char smem[];
    float*    sxl    = (float*)smem;                   // 64*SROW
    float*    sxr    = sxl + NPG * SROW;               // 64*SROW (later: out accum)
    float*    salpha = sxr + NPG * SROW;               // EPG*HH
    unsigned* smax   = (unsigned*)(salpha + EPG * HH); // 64*HH
    float*    sden   = (float*)(smax + NPG * HH);      // 64*HH
    float*    satt   = sden + NPG * HH;                // HC
    float*    seW    = satt + HC;                      // 8*HC

    const int tid = threadIdx.x;
    const int g   = blockIdx.x;
    const int nb  = g * NPG;
    const int eb  = g * EPG;

    // stage node features (vector global loads, scalar LDS stores due to pad)
    for (int q = tid; q < NPG * (HC / 4); q += 256) {
        int node = q >> 7;            // HC/4 = 128 quads per node
        int j4   = (q & 127) << 2;
        v4f a = *(const v4f*)(xl + (size_t)(nb + node) * HC + j4);
        v4f b = *(const v4f*)(xr + (size_t)(nb + node) * HC + j4);
        float* pl = sxl + node * SROW + j4;
        float* pr = sxr + node * SROW + j4;
#pragma unroll
        for (int i = 0; i < 4; ++i) { pl[i] = a[i]; pr[i] = b[i]; }
    }
    for (int j = tid; j < 8 * HC; j += 256) seW[j] = eW[j];
    for (int j = tid; j < HC; j += 256)     satt[j] = att[j];
    for (int j = tid; j < NPG * HH; j += 256) { smax[j] = 0u; sden[j] = 0.f; }
    __syncthreads();

    // pass 1: attention logits + per-(dst,head) running max
    for (int ei = tid; ei < EPG; ei += 256) {
        int s = esrc[eb + ei] - nb;
        int d = edst[eb + ei] - nb;
        int a = eattr[eb + ei];
        const float* pl = sxl + s * SROW;
        const float* pr = sxr + d * SROW;
        const float* pe = seW + a * HC;
#pragma unroll
        for (int h = 0; h < HH; ++h) {
            const int o = h * CC;
            float acc = 0.f;
            for (int c = 0; c < CC; ++c) {
                float v = pl[o + c] + pr[o + c] + pe[o + c];
                v = v > 0.f ? v : 0.2f * v;           // leaky_relu(0.2)
                acc += satt[o + c] * v;
            }
            salpha[ei * HH + h] = acc;
            atomicMax(&smax[d * HH + h], fenc(acc));
        }
    }
    __syncthreads();

    // pass 2: exp(alpha - max) + per-(dst,head) denom
    for (int ei = tid; ei < EPG; ei += 256) {
        int d = edst[eb + ei] - nb;
#pragma unroll
        for (int h = 0; h < HH; ++h) {
            float ea = __expf(salpha[ei * HH + h] - fdec(smax[d * HH + h]));
            salpha[ei * HH + h] = ea;
            atomicAdd(&sden[d * HH + h], ea);
        }
    }
    __syncthreads();

    // reuse sxr as output accumulator
    for (int q = tid; q < NPG * SROW; q += 256) sxr[q] = 0.f;
    __syncthreads();

    // pass 3: weighted aggregation into destination rows
    for (int ei = tid; ei < EPG; ei += 256) {
        int s = esrc[eb + ei] - nb;
        int d = edst[eb + ei] - nb;
        const float* pl = sxl + s * SROW;
        float*       po = sxr + d * SROW;
#pragma unroll
        for (int h = 0; h < HH; ++h) {
            float aa = salpha[ei * HH + h] / (sden[d * HH + h] + 1e-16f);
            const int o = h * CC;
            for (int c = 0; c < CC; ++c)
                atomicAdd(&po[o + c], aa * pl[o + c]);
        }
    }
    __syncthreads();

    // finalize: +bias, LayerNorm, ReLU (one thread per node)
    if (tid < NPG) {
        const float* po = sxr + tid * SROW;
        if (concat) {
            float sum = 0.f, ss = 0.f;
            for (int j = 0; j < HC; ++j) {
                float v = po[j] + bias[j];
                sum += v; ss += v * v;
            }
            float mu  = sum * (1.f / HC);
            float var = ss * (1.f / HC) - mu * mu;
            float rs  = rsqrtf(var + 1e-5f);
            for (int j = 0; j < HC; ++j) {
                float v = (po[j] + bias[j] - mu) * rs * lng[j] + lnb[j];
                v = v > 0.f ? v : 0.f;
                outbf[(size_t)(nb + tid) * HC + j] = (__bf16)v;
            }
        } else {
            float sum = 0.f, ss = 0.f;
            for (int c = 0; c < CC; ++c) {
                float v = (po[c] + po[CC + c] + po[2 * CC + c] + po[3 * CC + c]) * 0.25f + bias[c];
                sum += v; ss += v * v;
            }
            float mu  = sum * (1.f / CC);
            float var = ss * (1.f / CC) - mu * mu;
            float rs  = rsqrtf(var + 1e-5f);
            for (int c = 0; c < CC; ++c) {
                float v = (po[c] + po[CC + c] + po[2 * CC + c] + po[3 * CC + c]) * 0.25f + bias[c];
                v = (v - mu) * rs * lng[c] + lnb[c];
                v = v > 0.f ? v : 0.f;
                outf[(size_t)(nb + tid) * CC + c] = v;
            }
        }
    }
}

// ---- global mean pool over the 64 nodes of each graph ----
__global__ __launch_bounds__(128) void pool_kernel(
    const float* __restrict__ x2, float* __restrict__ gmean, __bf16* __restrict__ gbf)
{
    int c = threadIdx.x;      // 128 channels
    int g = blockIdx.x;
    float s = 0.f;
    for (int i = 0; i < NPG; ++i)
        s += x2[(size_t)(g * NPG + i) * CC + c];
    s *= (1.f / NPG);
    gmean[g * CC + c] = s;
    gbf[g * CC + c]   = (__bf16)s;
}

// ---- small MLP heads (128 hidden -> 1), mode 0=tanh, 1=sigmoid ----
__global__ __launch_bounds__(128) void head_kernel(
    const float* __restrict__ g, const float* __restrict__ W1,
    const float* __restrict__ b1, const float* __restrict__ W2,
    const float* __restrict__ b2, float* __restrict__ out, int mode)
{
    __shared__ float red[128];
    int t = threadIdx.x, gi = blockIdx.x;
    const float* gv = g + gi * CC;
    float acc = b1[t];
    for (int k = 0; k < CC; ++k) acc += gv[k] * W1[k * 128 + t];
    acc = acc > 0.f ? acc : 0.f;
    red[t] = acc * W2[t];
    __syncthreads();
    for (int s = 64; s > 0; s >>= 1) {
        if (t < s) red[t] += red[t + s];
        __syncthreads();
    }
    if (t == 0) {
        float v = red[0] + b2[0];
        out[gi] = mode ? (1.f / (1.f + __expf(-v))) : tanhf(v);
    }
}

// ---- transpose + f32 -> bf16 convert: in[K,Nc] -> out[Nc,K] ----
__global__ void cvtT_kernel(const float* __restrict__ in, __bf16* __restrict__ out,
                            int K, int Nc)
{
    size_t total = (size_t)K * Nc;
    for (size_t i = blockIdx.x * (size_t)blockDim.x + threadIdx.x; i < total;
         i += (size_t)gridDim.x * blockDim.x) {
        size_t k = i / Nc, n = i - k * Nc;
        out[n * K + k] = (__bf16)in[i];
    }
}

// ---- precompute eW[a, :] = edge_emb[a, :] @ We  (8 x HC) ----
__global__ void edgeW_kernel(const float* __restrict__ emb,
                             const float* __restrict__ We, float* __restrict__ out)
{
    int j = blockIdx.x * blockDim.x + threadIdx.x;
    if (j >= 8 * HC) return;
    int a = j >> 9, c = j & (HC - 1);
    float s = 0.f;
    for (int k = 0; k < ED; ++k) s += emb[a * ED + k] * We[k * HC + c];
    out[j] = s;
}

// ---- gather node embeddings -> bf16 [N, 64] ----
__global__ void gather_kernel(const float* __restrict__ node_emb,
                              const int* __restrict__ x_idx, __bf16* __restrict__ xbf)
{
    size_t total = (size_t)NN * ND;
    for (size_t i = blockIdx.x * (size_t)blockDim.x + threadIdx.x; i < total;
         i += (size_t)gridDim.x * blockDim.x) {
        size_t n = i >> 6, j = i & 63;
        xbf[i] = (__bf16)node_emb[(size_t)x_idx[n] * ND + j];
    }
}

extern "C" void kernel_launch(void* const* d_in, const int* in_sizes, int n_in,
                              void* d_out, int out_size, void* d_ws, size_t ws_size,
                              hipStream_t stream) {
    const int*   x_idx    = (const int*)d_in[0];
    const int*   esrc     = (const int*)d_in[1];
    const int*   edst     = (const int*)d_in[2];
    const int*   eattr    = (const int*)d_in[3];
    // d_in[4] = batch (regular structure; graph id == blockIdx)
    const float* node_emb = (const float*)d_in[5];
    const float* edge_emb = (const float*)d_in[6];
    const float* g1_Wl    = (const float*)d_in[7];
    const float* g1_bl    = (const float*)d_in[8];
    const float* g1_Wr    = (const float*)d_in[9];
    const float* g1_br    = (const float*)d_in[10];
    const float* g1_We    = (const float*)d_in[11];
    const float* g1_att   = (const float*)d_in[12];
    const float* g1_bias  = (const float*)d_in[13];
    const float* g2_Wl    = (const float*)d_in[14];
    const float* g2_bl    = (const float*)d_in[15];
    const float* g2_Wr    = (const float*)d_in[16];
    const float* g2_br    = (const float*)d_in[17];
    const float* g2_We    = (const float*)d_in[18];
    const float* g2_att   = (const float*)d_in[19];
    const float* g2_bias  = (const float*)d_in[20];
    const float* ln1_g    = (const float*)d_in[21];
    const float* ln1_b    = (const float*)d_in[22];
    const float* ln2_g    = (const float*)d_in[23];
    const float* ln2_b    = (const float*)d_in[24];
    const float* v_W1     = (const float*)d_in[25];
    const float* v_b1     = (const float*)d_in[26];
    const float* v_W2     = (const float*)d_in[27];
    const float* v_b2     = (const float*)d_in[28];
    const float* p_W1     = (const float*)d_in[29];
    const float* p_b1     = (const float*)d_in[30];
    const float* p_W2     = (const float*)d_in[31];
    const float* p_b2     = (const float*)d_in[32];
    const float* t_W1     = (const float*)d_in[33];
    const float* t_b1     = (const float*)d_in[34];
    const float* t_W2     = (const float*)d_in[35];
    const float* t_b2     = (const float*)d_in[36];
    const float* s_W1     = (const float*)d_in[37];
    const float* s_b1     = (const float*)d_in[38];
    const float* s_W2     = (const float*)d_in[39];
    const float* s_b2     = (const float*)d_in[40];

    // workspace carve (256B aligned)
    char* p = (char*)d_ws;
    auto alloc = [&](size_t bytes) -> void* {
        void* r = (void*)p;
        p += (bytes + 255) & ~(size_t)255;
        return r;
    };
    __bf16* xbf   = (__bf16*)alloc((size_t)NN * ND * 2);
    __bf16* W1lT  = (__bf16*)alloc((size_t)HC * ND * 2);
    __bf16* W1rT  = (__bf16*)alloc((size_t)HC * ND * 2);
    __bf16* W2lT  = (__bf16*)alloc((size_t)HC * HC * 2);
    __bf16* W2rT  = (__bf16*)alloc((size_t)HC * HC * 2);
    float*  eW1   = (float*)alloc((size_t)8 * HC * 4);
    float*  eW2   = (float*)alloc((size_t)8 * HC * 4);
    float*  xl    = (float*)alloc((size_t)NN * HC * 4);
    float*  xr    = (float*)alloc((size_t)NN * HC * 4);
    __bf16* h1bf  = (__bf16*)alloc((size_t)NN * HC * 2);
    float*  x2    = (float*)alloc((size_t)NN * CC * 4);
    float*  gmean = (float*)alloc((size_t)NUM_GRAPHS * CC * 4);
    __bf16* gbf   = (__bf16*)alloc((size_t)NUM_GRAPHS * CC * 2);
    __bf16* pW1T  = (__bf16*)alloc((size_t)256 * CC * 2);
    __bf16* pW2T  = (__bf16*)alloc((size_t)POLICY * 256 * 2);
    __bf16* hpbf  = (__bf16*)alloc((size_t)256 * 256 * 2);

    // ---- prep ----
    cvtT_kernel<<<64,  256, 0, stream>>>(g1_Wl, W1lT, ND, HC);
    cvtT_kernel<<<64,  256, 0, stream>>>(g1_Wr, W1rT, ND, HC);
    cvtT_kernel<<<256, 256, 0, stream>>>(g2_Wl, W2lT, HC, HC);
    cvtT_kernel<<<256, 256, 0, stream>>>(g2_Wr, W2rT, HC, HC);
    cvtT_kernel<<<64,  256, 0, stream>>>(p_W1,  pW1T, CC, 256);
    cvtT_kernel<<<512, 256, 0, stream>>>(p_W2,  pW2T, 256, POLICY);
    edgeW_kernel<<<16, 256, 0, stream>>>(edge_emb, g1_We, eW1);
    edgeW_kernel<<<16, 256, 0, stream>>>(edge_emb, g2_We, eW2);
    gather_kernel<<<1024, 256, 0, stream>>>(node_emb, x_idx, xbf);

    // LDS bytes for gatv2
    size_t smem = (size_t)(2 * NPG * SROW + EPG * HH + NPG * HH * 2 + HC + 8 * HC) * 4;

    // ---- layer 1: xl/xr = x @ W{l,r} + b  (M=NN, K=64, Nc=512) ----
    {
        int tiles  = (NN / 16) * (HC / 64);   // 8192 strip-tiles
        int blocks = tiles / 8;               // 1024
        gemm_bf16_wmma<<<blocks, 256, 0, stream>>>(xbf, W1lT, g1_bl, xl, nullptr, NN, HC, ND, 0);
        gemm_bf16_wmma<<<blocks, 256, 0, stream>>>(xbf, W1rT, g1_br, xr, nullptr, NN, HC, ND, 0);
    }
    gatv2_kernel<<<NUM_GRAPHS, 256, smem, stream>>>(
        xl, xr, eW1, esrc, edst, eattr, g1_att, g1_bias, ln1_g, ln1_b,
        1, nullptr, h1bf);

    // ---- layer 2 (reuse xl/xr buffers; K=512) ----
    {
        int tiles  = (NN / 16) * (HC / 64);
        int blocks = tiles / 8;
        gemm_bf16_wmma<<<blocks, 256, 0, stream>>>(h1bf, W2lT, g2_bl, xl, nullptr, NN, HC, HC, 0);
        gemm_bf16_wmma<<<blocks, 256, 0, stream>>>(h1bf, W2rT, g2_br, xr, nullptr, NN, HC, HC, 0);
    }
    gatv2_kernel<<<NUM_GRAPHS, 256, smem, stream>>>(
        xl, xr, eW2, esrc, edst, eattr, g2_att, g2_bias, ln2_g, ln2_b,
        0, x2, nullptr);

    // ---- pool + heads ----
    pool_kernel<<<NUM_GRAPHS, 128, 0, stream>>>(x2, gmean, gbf);

    float* out        = (float*)d_out;
    float* out_value  = out;
    float* out_policy = out + NUM_GRAPHS;
    float* out_tactic = out_policy + (size_t)NUM_GRAPHS * POLICY;
    float* out_strat  = out_tactic + NUM_GRAPHS;

    // policy: hp = relu(g @ pW1 + b1) ; policy = hp @ pW2 + b2
    {
        int tiles1  = (256 / 16) * (256 / 64);     // 64
        gemm_bf16_wmma<<<tiles1 / 8, 256, 0, stream>>>(gbf, pW1T, p_b1, nullptr, hpbf, 256, 256, CC, 1);
        int tiles2  = (256 / 16) * (POLICY / 64);  // 16*73 = 1168
        gemm_bf16_wmma<<<(tiles2 + 7) / 8, 256, 0, stream>>>(hpbf, pW2T, p_b2, out_policy, nullptr, 256, POLICY, 256, 0);
    }

    head_kernel<<<NUM_GRAPHS, 128, 0, stream>>>(gmean, v_W1, v_b1, v_W2, v_b2, out_value, 0);
    head_kernel<<<NUM_GRAPHS, 128, 0, stream>>>(gmean, t_W1, t_b1, t_W2, t_b2, out_tactic, 1);
    head_kernel<<<NUM_GRAPHS, 128, 0, stream>>>(gmean, s_W1, s_b1, s_W2, s_b2, out_strat, 1);
}
